// UMABL_2302102471281
// MI455X (gfx1250) — matrix operand order repaired and checked by reading, not compiled, not run on hardware
//
#include <hip/hip_runtime.h>

// ---------------------------------------------------------------------------
// MI455X (gfx1250) implementation.
//
// Roofline: ~10.5 GFLOP of 1x1-conv GEMMs vs ~300 MB of HBM traffic
// => memory-bound (~13 us floor @ 23.3 TB/s). All channel-mixing GEMMs use
// v_wmma_f32_16x16x32_bf16 (bf16 in, f32 accumulate); activations feeding
// GEMMs are stored pixel-major bf16 so a lane's B fragment is a single 32B
// contiguous load; weights are pre-packed into a contiguous-k (same as B) A-fragment
// lane layout. Bias + residual adds are fused into GEMM epilogues.
// The K-loop is fully unrolled (template NK) with all B fragments preloaded
// and A fragments double-buffered, so fragment loads stay in flight while
// the previous tile's WMMAs execute (no per-WMMA s_wait_loadcnt 0 stall).
// Everything (<100 MB) is L2-resident (192 MB), so tap amplification in
// attention / depthwise convs stays off HBM.
// ---------------------------------------------------------------------------

#define NP    25600     // total pixels: 4 * 80 * 80
#define NPB   6400      // pixels per batch image
#define WIDTH 80
#define CCH   128

typedef __attribute__((ext_vector_type(16))) __bf16 v16bf;
typedef __attribute__((ext_vector_type(8)))  float  v8f;

__device__ __forceinline__ unsigned short f2bf(float f) {
  union { float f; unsigned int u; } v; v.f = f;
  unsigned int u = v.u;
  unsigned int r = u + 0x7fffu + ((u >> 16) & 1u);   // round-to-nearest-even
  return (unsigned short)(r >> 16);
}

__device__ __forceinline__ float gelu_exact(float x) {
  return 0.5f * x * (1.0f + erff(x * 0.70710678118654752f));
}

// Virtual GEMM row -> actual weight/bias row.
// mode 0: qkv0 head-slice (heads 0..5 of q,k,v): vr in [0,288)
// mode 1: qkv1 head-slice (heads 6..7 of q,k,v): vr in [0,96)
// mode >=2: identity
__device__ __forceinline__ int map_row(int gr, int mode) {
  if (mode == 0) { int third = gr / 96; int sub = gr % 96;      return third * 128 + sub; }
  if (mode == 1) { int third = gr / 32; int sub = 96 + gr % 32; return third * 128 + sub; }
  return gr;
}

// ---------------------------------------------------------------------------
// Weight prep: f32 [OC][K] row-major -> bf16 WMMA A-fragment layout.
// Fragment storage: dst[((tile*nK + kk)*32 + lane)*16 + e]
// 16-bit A 16x32 layout used here (contiguous k, matches B): lane m = lane&15, half = lane>>4,
//   element e: k = kk*32 + half*16 + e
// Padded tiles (gr >= OCreal) are zero-filled so the GEMM can always run
// 4 fully-unrolled WMMA tiles per wave.
// ---------------------------------------------------------------------------
__global__ __launch_bounds__(256) void prep_w_kernel(
    const float* __restrict__ W, unsigned short* __restrict__ dst,
    int K, int nK, int nTilesPad, int OCreal, int mode)
{
  int t = blockIdx.x * 256 + threadIdx.x;
  int total = nTilesPad * nK * 512;
  if (t >= total) return;
  int e    = t & 15;
  int lane = (t >> 4) & 31;
  int rest = t >> 9;
  int kk   = rest % nK;
  int tile = rest / nK;
  int m    = lane & 15, half = lane >> 4;
  int v    = e >> 1;
  int k    = kk * 32 + half * 16 + e;
  int gr   = tile * 16 + m;
  float val = 0.f;
  if (gr < OCreal) val = W[(size_t)map_row(gr, mode) * K + k];
  dst[t] = f2bf(val);
}

// ---------------------------------------------------------------------------
// ln2d over channels. mode 0: write bf16 pixel-major [pg][128] (GEMM input).
//                     mode 1: write f32 NCHW (for depthwise-conv stage).
// ---------------------------------------------------------------------------
__global__ __launch_bounds__(256) void ln2d_kernel(
    const float* __restrict__ x, const float* __restrict__ w, const float* __restrict__ b,
    unsigned short* __restrict__ dst_bf, float* __restrict__ dst_f32, int mode)
{
  int pg = blockIdx.x * 256 + threadIdx.x;
  if (pg >= NP) return;
  int bb = pg / NPB, pin = pg % NPB;
  const float* xp = x + (size_t)bb * CCH * NPB + pin;
  float vals[CCH];
  float s = 0.f, s2 = 0.f;
#pragma unroll
  for (int c = 0; c < CCH; ++c) {
    float t = xp[(size_t)c * NPB];
    vals[c] = t; s += t; s2 += t * t;
  }
  float mu  = s * (1.0f / CCH);
  float var = s2 * (1.0f / CCH) - mu * mu;
  float rs  = rsqrtf(var + 1e-6f);
#pragma unroll
  for (int c = 0; c < CCH; ++c) {
    float y = (vals[c] - mu) * rs * w[c] + b[c];
    if (mode == 0) dst_bf[(size_t)pg * CCH + c] = f2bf(y);
    else           dst_f32[(size_t)bb * CCH * NPB + (size_t)c * NPB + pin] = y;
  }
}

// ---------------------------------------------------------------------------
// Generic 1x1-conv GEMM: out[gr][p] = bias[gr] + sum_k W[gr][k] * Xb[p][k]
// One wave computes a 64(oc) x 16(px) tile. B fragments for the whole K
// extent are preloaded (NK <= 4 -> 32 VGPRs); A fragments are double-
// buffered across the 4 oc tiles so loads overlap WMMA execution.
// All branches around the WMMAs are wave-uniform => EXEC all-1s.
// Epilogue modes:
//   0/1: qkv scatter -> f32 channel-major Q/K/V buffers [128][NP]
//   2:   residual: o0[nchw] = resid[nchw] + val      (f32 NCHW)
//   3:   f32 channel-major o0[gr*NP + p]             (msconvstar hidden h)
// ---------------------------------------------------------------------------
template <int NK>
__global__ __launch_bounds__(256) void gemm_wmma_kernel(
    const unsigned short* __restrict__ Xb, const unsigned short* __restrict__ WF,
    const float* __restrict__ bias, int K, int nTilesReal, int nTilesPad,
    int mode, float* __restrict__ o0, float* __restrict__ o1, float* __restrict__ o2,
    const float* __restrict__ resid)
{
  int wid  = (int)((blockIdx.x * 256u + threadIdx.x) >> 5);
  int lane = (int)(threadIdx.x & 31u);
  int nOcBlk = nTilesPad >> 2;
  int totalWaves = 1600 * nOcBlk;            // 1600 pixel tiles of 16
  if (wid >= totalWaves) return;             // wave-uniform
  int pxTile = wid % 1600;
  int ocBlk  = wid / 1600;
  int pxBase = pxTile << 4;
  int n = lane & 15, half = lane >> 4;
  int tile0 = ocBlk << 2;

  v8f acc[4];
#pragma unroll
  for (int j = 0; j < 4; ++j) {
    int tile = tile0 + j;
    bool valid = tile < nTilesReal;
#pragma unroll
    for (int r = 0; r < 8; ++r) {
      int gr = tile * 16 + r + 8 * half;     // C/D layout: m = r + 8*(lane>=16)
      acc[j][r] = valid ? bias[map_row(gr, mode)] : 0.f;
    }
  }

  // Preload ALL B fragments: lane (n, half) holds Xb[pxBase+n][kk*32+half*16 ..].
  const unsigned short* xrow = Xb + (size_t)(pxBase + n) * (size_t)K + (size_t)half * 16;
  v16bf bfr[NK];
#pragma unroll
  for (int kk = 0; kk < NK; ++kk)
    bfr[kk] = *(const v16bf*)(xrow + kk * 32);

  // A fragments, double-buffered across the 4 oc tiles.
  const unsigned short* wfbase = WF + (size_t)lane * 16;
  v16bf acur[NK], anxt[NK];
#pragma unroll
  for (int kk = 0; kk < NK; ++kk)
    acur[kk] = *(const v16bf*)(wfbase + (((size_t)tile0 * NK + kk) << 9));

#pragma unroll
  for (int j = 0; j < 4; ++j) {
    if (j < 3) {
#pragma unroll
      for (int kk = 0; kk < NK; ++kk)
        anxt[kk] = *(const v16bf*)(wfbase + ((((size_t)(tile0 + j + 1)) * NK + kk) << 9));
    }
#pragma unroll
    for (int kk = 0; kk < NK; ++kk)
      acc[j] = __builtin_amdgcn_wmma_f32_16x16x32_bf16(
          false, acur[kk], false, bfr[kk], (short)0, acc[j], false, false);
    if (j < 3) {
#pragma unroll
      for (int kk = 0; kk < NK; ++kk) acur[kk] = anxt[kk];
    }
  }

  int p = pxBase + n;
#pragma unroll
  for (int j = 0; j < 4; ++j) {
    int tile = tile0 + j;
    if (tile >= nTilesReal) continue;
#pragma unroll
    for (int r = 0; r < 8; ++r) {
      int gr = tile * 16 + r + 8 * half;
      float val = acc[j][r];
      if (mode <= 1) {
        int third, sub;
        if (mode == 0) { third = gr / 96; sub = gr % 96; }
        else           { third = gr / 32; sub = 96 + gr % 32; }
        float* dst = (third == 0) ? o0 : (third == 1) ? o1 : o2;
        dst[(size_t)sub * NP + p] = val;
      } else if (mode == 2) {
        int bb = p / NPB, pin = p % NPB;
        size_t idx = ((size_t)bb * CCH + gr) * NPB + pin;
        o0[idx] = resid[idx] + val;
      } else {
        o0[(size_t)gr * NP + p] = val;
      }
    }
  }
}

// ---------------------------------------------------------------------------
// Neighborhood attention. Q/K/V are f32 [row=head*16+d][pg]. Padding matches
// the reference: out-of-bounds neighbors contribute logit 0 (zero-padded k)
// and v = 0, but still participate in the softmax denominator.
// Templated on (KS,DIL) so logit arrays stay fully register-resident.
// ---------------------------------------------------------------------------
template <int KS, int DIL>
__device__ __forceinline__ void nattn_core(
    const float* __restrict__ Qb, const float* __restrict__ Kb,
    const float* __restrict__ Vb, unsigned short* __restrict__ out,
    int head, int pg)
{
  const int CTR = KS / 2;
  int bb = pg / NPB, pin = pg % NPB;
  int yy = pin / WIDTH, xx = pin % WIDTH;
  float q[16];
#pragma unroll
  for (int d = 0; d < 16; ++d)
    q[d] = Qb[(size_t)(head * 16 + d) * NP + pg] * 0.25f;   // HD^-0.5

  float lg[KS * KS];
#pragma unroll
  for (int i = 0; i < KS; ++i) {
#pragma unroll
    for (int j = 0; j < KS; ++j) {
      int ny = yy + (i - CTR) * DIL, nx = xx + (j - CTR) * DIL;
      float acc = 0.f;
      if (ny >= 0 && ny < WIDTH && nx >= 0 && nx < WIDTH) {
        int npg = bb * NPB + ny * WIDTH + nx;
#pragma unroll
        for (int d = 0; d < 16; ++d)
          acc += q[d] * Kb[(size_t)(head * 16 + d) * NP + npg];
      }
      lg[i * KS + j] = acc;
    }
  }
  float mx = lg[0];
#pragma unroll
  for (int t = 1; t < KS * KS; ++t) mx = fmaxf(mx, lg[t]);
  float den = 0.f;
#pragma unroll
  for (int t = 0; t < KS * KS; ++t) { lg[t] = expf(lg[t] - mx); den += lg[t]; }

  float o[16];
#pragma unroll
  for (int d = 0; d < 16; ++d) o[d] = 0.f;
#pragma unroll
  for (int i = 0; i < KS; ++i) {
#pragma unroll
    for (int j = 0; j < KS; ++j) {
      int ny = yy + (i - CTR) * DIL, nx = xx + (j - CTR) * DIL;
      if (ny >= 0 && ny < WIDTH && nx >= 0 && nx < WIDTH) {
        int npg = bb * NPB + ny * WIDTH + nx;
        float wgt = lg[i * KS + j];
#pragma unroll
        for (int d = 0; d < 16; ++d)
          o[d] += wgt * Vb[(size_t)(head * 16 + d) * NP + npg];
      }
    }
  }
  float inv = 1.f / den;
#pragma unroll
  for (int d = 0; d < 16; ++d)
    out[(size_t)pg * CCH + head * 16 + d] = f2bf(o[d] * inv);   // bf16 pixel-major
}

__global__ __launch_bounds__(256) void nattn_kernel(
    const float* __restrict__ Qb, const float* __restrict__ Kb,
    const float* __restrict__ Vb, unsigned short* __restrict__ out)
{
  int t = blockIdx.x * 256 + threadIdx.x;
  if (t >= 8 * NP) return;
  int pg = t % NP;
  int head = t / NP;
  if (head < 6) nattn_core<3, 1>(Qb, Kb, Vb, out, head, pg);
  else          nattn_core<5, 2>(Qb, Kb, Vb, out, head, pg);
}

// ---------------------------------------------------------------------------
// msconvstar middle: h (f32 [192][NP]) -> depthwise (1x1 / 3x3 / 5x5 on the
// three 64-ch groups) + residual + star gate gelu(x1)*x2 -> bf16 [pg][96].
// ---------------------------------------------------------------------------
__device__ float dw_apply(const float* __restrict__ h, int ch, int bb, int yy, int xx,
                          const float* __restrict__ w0, const float* __restrict__ b0,
                          const float* __restrict__ w1, const float* __restrict__ b1,
                          const float* __restrict__ w2, const float* __restrict__ b2)
{
  const float* hc = h + (size_t)ch * NP + (size_t)bb * NPB;
  if (ch < 64) return w0[ch] * hc[yy * WIDTH + xx] + b0[ch];
  int k; const float* wp; float acc;
  if (ch < 128) { k = 3; wp = w1 + (ch - 64)  * 9;  acc = b1[ch - 64];  }
  else          { k = 5; wp = w2 + (ch - 128) * 25; acc = b2[ch - 128]; }
  int ctr = k / 2;
  for (int i = 0; i < k; ++i) {
    int ny = yy + i - ctr;
    if (ny < 0 || ny >= WIDTH) continue;
    for (int j = 0; j < k; ++j) {
      int nx = xx + j - ctr;
      if (nx < 0 || nx >= WIDTH) continue;
      acc += wp[i * k + j] * hc[ny * WIDTH + nx];
    }
  }
  return acc;
}

__global__ __launch_bounds__(256) void dwstar_kernel(
    const float* __restrict__ h,
    const float* __restrict__ w0, const float* __restrict__ b0,
    const float* __restrict__ w1, const float* __restrict__ b1,
    const float* __restrict__ w2, const float* __restrict__ b2,
    unsigned short* __restrict__ out)
{
  int t = blockIdx.x * 256 + threadIdx.x;
  if (t >= 96 * NP) return;
  int pg = t % NP, c = t / NP;
  int bb = pg / NPB, pin = pg % NPB;
  int yy = pin / WIDTH, xx = pin % WIDTH;
  float y1 = h[(size_t)c * NP + pg]        + dw_apply(h, c,      bb, yy, xx, w0, b0, w1, b1, w2, b2);
  float y2 = h[(size_t)(c + 96) * NP + pg] + dw_apply(h, c + 96, bb, yy, xx, w0, b0, w1, b1, w2, b2);
  out[(size_t)pg * 96 + c] = f2bf(gelu_exact(y1) * y2);
}

// ---------------------------------------------------------------------------
// cascaded_sparse depthwise 3x3 (dilated) + GELU.
// src is f32 NCHW. mode 0: f32 NCHW out.  mode 1: bf16 pixel-major [pg][128].
// ---------------------------------------------------------------------------
__global__ __launch_bounds__(256) void dw3_gelu_kernel(
    const float* __restrict__ src, const float* __restrict__ w, const float* __restrict__ b,
    unsigned short* __restrict__ dst_bf, float* __restrict__ dst_f32, int dil, int mode)
{
  int t = blockIdx.x * 256 + threadIdx.x;
  if (t >= CCH * NP) return;
  int pg = t % NP, c = t / NP;
  int bb = pg / NPB, pin = pg % NPB;
  int yy = pin / WIDTH, xx = pin % WIDTH;
  const float* sc = src + ((size_t)bb * CCH + c) * NPB;
  const float* wp = w + c * 9;
  float acc = b[c];
#pragma unroll
  for (int i = 0; i < 3; ++i) {
    int ny = yy + (i - 1) * dil;
    if (ny < 0 || ny >= WIDTH) continue;
#pragma unroll
    for (int j = 0; j < 3; ++j) {
      int nx = xx + (j - 1) * dil;
      if (nx < 0 || nx >= WIDTH) continue;
      acc += wp[i * 3 + j] * sc[ny * WIDTH + nx];
    }
  }
  float y = gelu_exact(acc);
  if (mode == 0) dst_f32[((size_t)bb * CCH + c) * NPB + pin] = y;
  else           dst_bf[(size_t)pg * CCH + c] = f2bf(y);
}

// ---------------------------------------------------------------------------
// Host orchestration
// ---------------------------------------------------------------------------
extern "C" void kernel_launch(void* const* d_in, const int* in_sizes, int n_in,
                              void* d_out, int out_size, void* d_ws, size_t ws_size,
                              hipStream_t stream)
{
  (void)in_sizes; (void)n_in; (void)out_size; (void)ws_size;
  const float* x      = (const float*)d_in[0];
  const float* n1_w = (const float*)d_in[1];  const float* n1_b = (const float*)d_in[2];
  const float* n2_w = (const float*)d_in[3];  const float* n2_b = (const float*)d_in[4];
  const float* n3_w = (const float*)d_in[5];  const float* n3_b = (const float*)d_in[6];
  const float* n4_w = (const float*)d_in[7];  const float* n4_b = (const float*)d_in[8];
  const float* qkv0_w = (const float*)d_in[9];  const float* qkv0_b = (const float*)d_in[10];
  const float* qkv1_w = (const float*)d_in[11]; const float* qkv1_b = (const float*)d_in[12];
  const float* proj_w = (const float*)d_in[13]; const float* proj_b = (const float*)d_in[14];
  const float* m1_fc1_w = (const float*)d_in[15]; const float* m1_fc1_b = (const float*)d_in[16];
  const float* m1_dw0_w = (const float*)d_in[17]; const float* m1_dw0_b = (const float*)d_in[18];
  const float* m1_dw1_w = (const float*)d_in[19]; const float* m1_dw1_b = (const float*)d_in[20];
  const float* m1_dw2_w = (const float*)d_in[21]; const float* m1_dw2_b = (const float*)d_in[22];
  const float* m1_fc2_w = (const float*)d_in[23]; const float* m1_fc2_b = (const float*)d_in[24];
  const float* m2_fc1_w = (const float*)d_in[25]; const float* m2_fc1_b = (const float*)d_in[26];
  const float* m2_dw0_w = (const float*)d_in[27]; const float* m2_dw0_b = (const float*)d_in[28];
  const float* m2_dw1_w = (const float*)d_in[29]; const float* m2_dw1_b = (const float*)d_in[30];
  const float* m2_dw2_w = (const float*)d_in[31]; const float* m2_dw2_b = (const float*)d_in[32];
  const float* m2_fc2_w = (const float*)d_in[33]; const float* m2_fc2_b = (const float*)d_in[34];
  const float* sma_dw0_w = (const float*)d_in[35]; const float* sma_dw0_b = (const float*)d_in[36];
  const float* sma_dw1_w = (const float*)d_in[37]; const float* sma_dw1_b = (const float*)d_in[38];
  const float* sma_pw_w  = (const float*)d_in[39]; const float* sma_pw_b  = (const float*)d_in[40];

  float* out = (float*)d_out;           // residual stream lives in d_out
  char*  ws  = (char*)d_ws;

  // ---- workspace layout (all offsets 256B aligned) ----
  const size_t SZ_BF   = (size_t)NP * CCH * 2;   //  6,553,600  bf16 pixel-major buffer
  const size_t SZ_CH32 = (size_t)CCH * NP * 4;   // 13,107,200  one f32 channel-major tensor
  const size_t OFF_BF  = 0;
  const size_t OFF_BIG = SZ_BF;                  // 39.3 MB region, reused per stage
  const size_t OFF_WF  = OFF_BIG + 3 * SZ_CH32;  // weight fragments (~320 KB)

  unsigned short* BF = (unsigned short*)(ws + OFF_BF);
  float* Qb = (float*)(ws + OFF_BIG);
  float* Kb = (float*)(ws + OFF_BIG + SZ_CH32);
  float* Vb = (float*)(ws + OFF_BIG + 2 * SZ_CH32);
  float* Hb = (float*)(ws + OFF_BIG);                    // 192*NP*4 = 19.7 MB
  float* L3 = (float*)(ws + OFF_BIG);                    // ln3 output, f32 NCHW
  float* CT = (float*)(ws + OFF_BIG + SZ_CH32);          // cascade temp, f32 NCHW

  size_t wfOff = OFF_WF;
  auto wf_alloc = [&](int nTp, int nK) { size_t o = wfOff; wfOff += (size_t)nTp * nK * 1024; return o; };
  size_t wf_qkv0 = wf_alloc(20, 4);
  size_t wf_qkv1 = wf_alloc(8, 4);
  size_t wf_proj = wf_alloc(8, 4);
  size_t wf_fc1a = wf_alloc(12, 4);
  size_t wf_fc2a = wf_alloc(8, 3);
  size_t wf_pw   = wf_alloc(8, 4);
  size_t wf_fc1b = wf_alloc(12, 4);
  size_t wf_fc2b = wf_alloc(8, 3);

  auto prep = [&](const float* W, size_t off, int K, int nK, int nTp, int OCr, int mode) {
    int total = nTp * nK * 512;
    prep_w_kernel<<<(total + 255) / 256, 256, 0, stream>>>(
        W, (unsigned short*)(ws + off), K, nK, nTp, OCr, mode);
  };
  auto gemm = [&](const unsigned short* Xb, size_t wfo, const float* bias, int K, int nK,
                  int nTr, int nTp, int mode, float* o0, float* o1, float* o2, const float* resid) {
    long waves = 1600L * (nTp >> 2);
    int blocks = (int)((waves * 32 + 255) / 256);
    if (nK == 4)
      gemm_wmma_kernel<4><<<blocks, 256, 0, stream>>>(
          Xb, (const unsigned short*)(ws + wfo), bias, K, nTr, nTp, mode, o0, o1, o2, resid);
    else
      gemm_wmma_kernel<3><<<blocks, 256, 0, stream>>>(
          Xb, (const unsigned short*)(ws + wfo), bias, K, nTr, nTp, mode, o0, o1, o2, resid);
  };

  // ---- pre-pack all weight matrices into WMMA A-fragment layout ----
  prep(qkv0_w,   wf_qkv0, 128, 4, 20, 288, 0);   // heads 0..5 of q,k,v
  prep(qkv1_w,   wf_qkv1, 128, 4,  8,  96, 1);   // heads 6..7 of q,k,v
  prep(proj_w,   wf_proj, 128, 4,  8, 128, 2);
  prep(m1_fc1_w, wf_fc1a, 128, 4, 12, 192, 2);
  prep(m1_fc2_w, wf_fc2a,  96, 3,  8, 128, 2);
  prep(sma_pw_w, wf_pw,   128, 4,  8, 128, 2);
  prep(m2_fc1_w, wf_fc1b, 128, 4, 12, 192, 2);
  prep(m2_fc2_w, wf_fc2b,  96, 3,  8, 128, 2);

  const int LNB = (NP + 255) / 256;              // 100 blocks
  const int NAB = (8 * NP + 255) / 256;          // attention
  const int DSB = (96 * NP + 255) / 256;         // dwstar
  const int DCB = (CCH * NP + 255) / 256;        // cascade dw

  // ---- stage 1: x += neighborhood_attn(ln1(x)) ----
  ln2d_kernel<<<LNB, 256, 0, stream>>>(x, n1_w, n1_b, BF, (float*)nullptr, 0);
  gemm(BF, wf_qkv0, qkv0_b, 128, 4, 18, 20, 0, Qb, Kb, Vb, nullptr);
  gemm(BF, wf_qkv1, qkv1_b, 128, 4,  6,  8, 1, Qb, Kb, Vb, nullptr);
  nattn_kernel<<<NAB, 256, 0, stream>>>(Qb, Kb, Vb, BF);
  gemm(BF, wf_proj, proj_b, 128, 4, 8, 8, 2, out, nullptr, nullptr, x);

  // ---- stage 2: x += msconvstar(ln2(x))  [m1] ----
  ln2d_kernel<<<LNB, 256, 0, stream>>>(out, n2_w, n2_b, BF, (float*)nullptr, 0);
  gemm(BF, wf_fc1a, m1_fc1_b, 128, 4, 12, 12, 3, Hb, nullptr, nullptr, nullptr);
  dwstar_kernel<<<DSB, 256, 0, stream>>>(Hb, m1_dw0_w, m1_dw0_b, m1_dw1_w, m1_dw1_b,
                                         m1_dw2_w, m1_dw2_b, BF);
  gemm(BF, wf_fc2a, m1_fc2_b, 96, 3, 8, 8, 2, out, nullptr, nullptr, out);

  // ---- stage 3: x += cascaded_sparse(ln3(x)) ----
  ln2d_kernel<<<LNB, 256, 0, stream>>>(out, n3_w, n3_b, (unsigned short*)nullptr, L3, 1);
  dw3_gelu_kernel<<<DCB, 256, 0, stream>>>(L3, sma_dw0_w, sma_dw0_b,
                                           (unsigned short*)nullptr, CT, 1, 0);
  dw3_gelu_kernel<<<DCB, 256, 0, stream>>>(CT, sma_dw1_w, sma_dw1_b,
                                           BF, (float*)nullptr, 2, 1);
  gemm(BF, wf_pw, sma_pw_b, 128, 4, 8, 8, 2, out, nullptr, nullptr, out);

  // ---- stage 4: x += msconvstar(ln4(x))  [m2] ----
  ln2d_kernel<<<LNB, 256, 0, stream>>>(out, n4_w, n4_b, BF, (float*)nullptr, 0);
  gemm(BF, wf_fc1b, m2_fc1_b, 128, 4, 12, 12, 3, Hb, nullptr, nullptr, nullptr);
  dwstar_kernel<<<DSB, 256, 0, stream>>>(Hb, m2_dw0_w, m2_dw0_b, m2_dw1_w, m2_dw1_b,
                                         m2_dw2_w, m2_dw2_b, BF);
  gemm(BF, wf_fc2b, m2_fc2_b, 96, 3, 8, 8, 2, out, nullptr, nullptr, out);
}
